// negative_log_partial_likelihood_55379308315381
// MI455X (gfx1250) — compile-verified, matrix operand order checked by reading, not run on hardware
//
#include <hip/hip_runtime.h>
#include <hip/hip_bf16.h>
#include <math.h>

typedef __attribute__((ext_vector_type(16))) _Float16       v16h;
typedef __attribute__((ext_vector_type(8)))  _Float16       v8h;
typedef __attribute__((ext_vector_type(8)))  float          v8f;
typedef __attribute__((ext_vector_type(8)))  unsigned short v8us;

#define N_MAX          8192
#define TILE_K         32
#define BLOCK_THREADS  256
#define WAVES_PER_BLK  (BLOCK_THREADS / 32)

// ---------------------------------------------------------------------------
// Kernel 1: e[j] = (f16) exp(risk[j]);  t16[j] = (u16) os_time[j]
// ---------------------------------------------------------------------------
__global__ void cox_prep(const float* __restrict__ risk,
                         const int*   __restrict__ os_time,
                         _Float16*       __restrict__ e_half,
                         unsigned short* __restrict__ t16,
                         int n) {
  int i = blockIdx.x * blockDim.x + threadIdx.x;
  if (i < n) {
    e_half[i] = (_Float16)__expf(risk[i]);
    t16[i]    = (unsigned short)os_time[i];
  }
}

// ---------------------------------------------------------------------------
// Kernel 2: risk_set_sum[i] = sum_j (t[j] >= t[i]) * e[j]  via WMMA.
// A fragment = masked exp values (documented 16-bit A 16x32 layout),
// B fragment = all ones (layout-independent), f32 accumulator.
// Branch-free software-pipelined K loop (last chunk peeled) so ds_load
// latency hides under the WMMA, which co-executes with the VALU selects.
// ---------------------------------------------------------------------------
__global__ __launch_bounds__(BLOCK_THREADS)
void cox_rss_wmma(const _Float16*       __restrict__ e_half,
                  const unsigned short* __restrict__ t16,
                  float*                __restrict__ rss,
                  int n) {
  __shared__ _Float16       e_lds[N_MAX];
  __shared__ unsigned short t_lds[N_MAX];

  // Cooperative stage of the full vectors into LDS (16B vectors -> b128).
  {
    int nclamp = (n < N_MAX) ? n : N_MAX;
    int nvec   = nclamp >> 3;                     // 8 halfs / 8 u16 = 16 bytes
    const uint4* eg = (const uint4*)e_half;
    const uint4* tg = (const uint4*)t16;
    uint4* el = (uint4*)e_lds;
    uint4* tl = (uint4*)t_lds;
    for (int v = threadIdx.x; v < nvec; v += blockDim.x) {
      el[v] = eg[v];
      tl[v] = tg[v];
    }
  }
  __syncthreads();

  const int lane    = threadIdx.x & 31;
  const int wave    = threadIdx.x >> 5;
  const int laneRow = lane & 15;      // M row owned by this lane (A frag); N column in C frag
  const int laneHi  = lane >> 4;      // 0: K{0..7,16..23} / C rows 0..7
                                      // 1: K{8..15,24..31} / C rows 8..15
  const int klo     = laneHi << 3;

  v16h bOne;
#pragma unroll
  for (int e = 0; e < 16; ++e) bOne[e] = (_Float16)1.0f;

  const int nTiles     = n >> 4;
  const int totalWaves = gridDim.x * WAVES_PER_BLK;

  for (int tile = blockIdx.x * WAVES_PER_BLK + wave; tile < nTiles;
       tile += totalWaves) {
    const int rowBase = tile << 4;
    const unsigned short ti = t_lds[rowBase + laneRow];

    // ---- software pipeline prologue: load chunk 0 ----
    v8h  e0 = *(const v8h*) &e_lds[klo];
    v8h  e1 = *(const v8h*) &e_lds[klo + 16];
    v8us t0 = *(const v8us*)&t_lds[klo];
    v8us t1 = *(const v8us*)&t_lds[klo + 16];

    v8f acc = {};

    // Steady state: unconditional prefetch of chunk k0+32 before the WMMA
    // on chunk k0 (n is a multiple of TILE_K and >= TILE_K).
#pragma unroll 2
    for (int k0 = 0; k0 < n - TILE_K; k0 += TILE_K) {
      const int j0 = k0 + TILE_K + klo;  // 16B aligned
      const int j1 = j0 + 16;
      v8h  ne0 = *(const v8h*) &e_lds[j0];
      v8h  ne1 = *(const v8h*) &e_lds[j1];
      v8us nt0 = *(const v8us*)&t_lds[j0];
      v8us nt1 = *(const v8us*)&t_lds[j1];

      v16h a;
#pragma unroll
      for (int e = 0; e < 8; ++e) {
        a[e]     = (t0[e] >= ti) ? e0[e] : (_Float16)0.0f;
        a[e + 8] = (t1[e] >= ti) ? e1[e] : (_Float16)0.0f;
      }

      // D = A x 1 + C : every output column holds sum_k A[m,k]
      acc = __builtin_amdgcn_wmma_f32_16x16x32_f16(
          /*neg_a=*/false, a, /*neg_b=*/false, bOne,
          /*c_mod=*/(short)0, acc, /*reuse_a=*/false, /*reuse_b=*/false);

      e0 = ne0; e1 = ne1; t0 = nt0; t1 = nt1;
    }

    // ---- epilogue: last chunk ----
    {
      v16h a;
#pragma unroll
      for (int e = 0; e < 8; ++e) {
        a[e]     = (t0[e] >= ti) ? e0[e] : (_Float16)0.0f;
        a[e + 8] = (t1[e] >= ti) ? e1[e] : (_Float16)0.0f;
      }
      acc = __builtin_amdgcn_wmma_f32_16x16x32_f16(
          false, a, false, bOne, (short)0, acc, false, false);
    }

    // C/D f32 layout: lane 0 VGPR v = row v; lane 16 VGPR v = row 8+v.
    if (laneRow == 0) {
      const int r = rowBase + (laneHi << 3);
#pragma unroll
      for (int v = 0; v < 8; ++v) rss[r + v] = acc[v];
    }
  }
}

// ---------------------------------------------------------------------------
// Kernel 3: loss = -sum((theta - log(rss)) * os) / sum(os)
// ---------------------------------------------------------------------------
__global__ __launch_bounds__(256)
void cox_finalize(const float* __restrict__ risk,
                  const int*   __restrict__ os,
                  const float* __restrict__ rss,
                  float*       __restrict__ out,
                  int n) {
  __shared__ float snum[256];
  __shared__ float sden[256];
  float num = 0.0f, den = 0.0f;
  for (int i = threadIdx.x; i < n; i += blockDim.x) {
    float ev = (float)os[i];
    num += (risk[i] - logf(rss[i])) * ev;
    den += ev;
  }
  snum[threadIdx.x] = num;
  sden[threadIdx.x] = den;
  __syncthreads();
  for (int s = 128; s > 0; s >>= 1) {
    if ((int)threadIdx.x < s) {
      snum[threadIdx.x] += snum[threadIdx.x + s];
      sden[threadIdx.x] += sden[threadIdx.x + s];
    }
    __syncthreads();
  }
  if (threadIdx.x == 0) out[0] = -snum[0] / sden[0];
}

// ---------------------------------------------------------------------------
extern "C" void kernel_launch(void* const* d_in, const int* in_sizes, int n_in,
                              void* d_out, int out_size, void* d_ws,
                              size_t ws_size, hipStream_t stream) {
  const float* risk    = (const float*)d_in[0];
  const int*   os      = (const int*)  d_in[1];
  const int*   os_time = (const int*)  d_in[2];
  const int    n       = in_sizes[0];           // 8192

  char* ws = (char*)d_ws;
  _Float16*       e_half = (_Float16*)ws;                          // n*2 bytes
  unsigned short* t16    = (unsigned short*)(ws + (size_t)n * 2);  // n*2 bytes
  float*          rss    = (float*)(ws + (size_t)n * 4);           // n*4 bytes

  cox_prep<<<(n + 255) / 256, 256, 0, stream>>>(risk, os_time, e_half, t16, n);

  const int nTiles = n / 16;                                       // 512
  const int blocks = (nTiles + WAVES_PER_BLK - 1) / WAVES_PER_BLK; // 64
  cox_rss_wmma<<<blocks, BLOCK_THREADS, 0, stream>>>(e_half, t16, rss, n);

  cox_finalize<<<1, 256, 0, stream>>>(risk, os, rss, (float*)d_out, n);
}
